// Generator_64450279243852
// MI455X (gfx1250) — matrix operand order repaired and checked by reading, not compiled, not run on hardware
//
#include <hip/hip_runtime.h>

// ---------------------------------------------------------------------------
// Problem constants (match reference)
// ---------------------------------------------------------------------------
#define B_TOT     2048
#define LABEL_DIM 16
#define NOISE_DIM 100
#define SEQ_DIM   32
#define TSTEPS    128     // MAX_SEQ_LEN
#define HID       512
#define NLAYERS   4
#define GATE4     2048    // 4*HID (W_ih row count per layer)

#define ROWS      32      // batch rows per workgroup
#define NWAVES    16      // waves per workgroup (512 threads)
#define LDS_STRIDE (HID + 8)

// Fragment-packed weight geometry:
//   block = (layer*3 + gate)*32 + ntile        (gate: 0=i, 1=g, 2=o)
//   inside block: [kk(16)][lane(32)][16 bf16]  -> 16 KB per block, each
//   kk-slab is a 1 KB fully-coalesced chunk (lane*32B contiguous).
#define BLK_ELEMS (16 * 32 * 16)   // 8192 bf16 per block

typedef __attribute__((ext_vector_type(16))) __bf16       v16bf;
typedef __attribute__((ext_vector_type(8)))  float        v8f;
typedef __attribute__((ext_vector_type(4)))  unsigned int u32x4;

static __device__ inline v8f wmma_bf16(v16bf a, v16bf b, v8f c) {
  return __builtin_amdgcn_wmma_f32_16x16x32_bf16(
      /*neg_a=*/false, a, /*neg_b=*/false, b,
      /*c_mod=*/(short)0, c, /*reuse_a=*/false, /*reuse_b=*/false);
}

// Fast transcendentals on v_exp_f32 / v_rcp_f32 (saturate correctly at +-inf)
static __device__ inline float fast_sigmoid(float x) {
  return __builtin_amdgcn_rcpf(1.0f + __expf(-x));
}
static __device__ inline float fast_tanh(float x) {
  return 1.0f - 2.0f * __builtin_amdgcn_rcpf(1.0f + __expf(2.0f * x));
}

// ---------------------------------------------------------------------------
// Kernel 1: repack weights into fragment-contiguous bf16 layout,
//           fuse b_ih+b_hh, repack W_out the same way.
// ---------------------------------------------------------------------------
__global__ void repack_kernel(const float* __restrict__ W_ih,
                              const float* __restrict__ b_ih,
                              const float* __restrict__ b_hh,
                              const float* __restrict__ W_out,
                              __bf16* __restrict__ Wpack,   // 384 blocks
                              float*  __restrict__ bcomb,   // [4][2048]
                              __bf16* __restrict__ Woutpack) { // 2 blocks
  const int i = blockIdx.x * blockDim.x + threadIdx.x;

  // --- W_ih (i,g,o gates only): 4*3*32 blocks * 16 kk * 32 lanes chunks ---
  if (i < NLAYERS * 3 * 32 * 16 * 32) {
    const int lane = i & 31;
    const int kk   = (i >> 5) & 15;
    const int blk  = i >> 9;
    const int nt   = blk & 31;
    const int lg   = blk >> 5;       // layer*3 + gate
    const int g    = lg % 3;
    const int l    = lg / 3;
    const int l16  = lane & 15;
    const int half = lane >> 4;
    const int gbase = (g == 0) ? 0 : (g == 1 ? 1024 : 1536);
    const int row  = gbase + nt * 16 + l16;
    const float* src = W_ih + ((size_t)l * GATE4 + row) * HID + kk * 32 + half * 8;
    __bf16* dst = Wpack + ((size_t)blk * 16 + kk) * (32 * 16) + lane * 16;
    #pragma unroll
    for (int e = 0; e < 16; ++e) dst[e] = (__bf16)src[e < 8 ? e : e + 8];
  }
  // --- fused bias ---
  if (i < NLAYERS * GATE4) bcomb[i] = b_ih[i] + b_hh[i];
  // --- W_out: 2 blocks ---
  if (i < 2 * 16 * 32) {
    const int lane = i & 31;
    const int kk   = (i >> 5) & 15;
    const int nt   = i >> 9;
    const int l16  = lane & 15;
    const int half = lane >> 4;
    const float* src = W_out + (size_t)(nt * 16 + l16) * HID + kk * 32 + half * 8;
    __bf16* dst = Woutpack + ((size_t)nt * 16 + kk) * (32 * 16) + lane * 16;
    #pragma unroll
    for (int e = 0; e < 16; ++e) dst[e] = (__bf16)src[e < 8 ? e : e + 8];
  }
}

// ---------------------------------------------------------------------------
// Kernel 2: conditioning front-end -> x0 (bf16, [B, HID])
// ---------------------------------------------------------------------------
__global__ void frontend_kernel(const float* __restrict__ label,
                                const float* __restrict__ noise,
                                const int*   __restrict__ length,
                                const float* __restrict__ W_label,
                                const float* __restrict__ b_label,
                                const float* __restrict__ W_noise,
                                const float* __restrict__ b_noise,
                                const float* __restrict__ W_len,
                                const float* __restrict__ b_len,
                                const float* __restrict__ W_comb,
                                const float* __restrict__ b_comb,
                                __bf16* __restrict__ x0) {
  __shared__ float comb[3 * 128];
  const int b   = blockIdx.x;
  const int tid = threadIdx.x;   // 128 threads

  {
    float s = b_label[tid];
    const float* lb = label + (size_t)b * LABEL_DIM;
    const float* wr = W_label + (size_t)tid * LABEL_DIM;
    #pragma unroll
    for (int k = 0; k < LABEL_DIM; ++k) s += lb[k] * wr[k];
    comb[tid] = fmaxf(s, 0.0f);

    s = b_noise[tid];
    const float* nz = noise + (size_t)b * NOISE_DIM;
    const float* wn = W_noise + (size_t)tid * NOISE_DIM;
    for (int k = 0; k < NOISE_DIM; ++k) s += nz[k] * wn[k];
    comb[128 + tid] = fmaxf(s, 0.0f);

    const int li = length[b] - 1;   // one_hot @ W_len.T == column gather
    comb[256 + tid] = fmaxf(W_len[(size_t)tid * TSTEPS + li] + b_len[tid], 0.0f);
  }
  __syncthreads();

  for (int j = tid; j < HID; j += 128) {
    float s = b_comb[j];
    const float* w = W_comb + (size_t)j * (3 * 128);
    for (int k = 0; k < 3 * 128; ++k) s += comb[k] * w[k];
    x0[(size_t)b * HID + j] = (__bf16)fmaxf(s, 0.0f);
  }
}

// ---------------------------------------------------------------------------
// Kernel 3: persistent per-batch-tile LSTM chain (WMMA bf16, f-gate skipped)
// ---------------------------------------------------------------------------
__global__ __launch_bounds__(NWAVES * 32, 1)
void lstm_gen_kernel(const __bf16* __restrict__ Wpack,    // 384 packed blocks
                     const float*  __restrict__ bc,       // [4][2048] fused bias
                     const __bf16* __restrict__ Woutpack, // 2 packed blocks
                     const float*  __restrict__ b_out,    // [32]
                     const __bf16* __restrict__ x0,       // [B][512] bf16
                     float*        __restrict__ out) {    // [B][128][32] f32
  __shared__ __align__(16) __bf16 hs[2][ROWS][LDS_STRIDE];

  const int tid  = threadIdx.x;
  const int wave = tid >> 5;
  const int lane = tid & 31;
  const int l16  = lane & 15;
  const int half = lane >> 4;
  const int wg   = blockIdx.x;

  // Preload this wave's gate biases (nt = wave and wave+16, fixed forever).
  float biasv[NLAYERS][2][3];
  #pragma unroll
  for (int l = 0; l < NLAYERS; ++l) {
    #pragma unroll
    for (int j = 0; j < 2; ++j) {
      const int col = (wave + j * 16) * 16 + l16;
      biasv[l][j][0] = bc[l * GATE4 + col];          // i
      biasv[l][j][1] = bc[l * GATE4 + 1024 + col];   // g
      biasv[l][j][2] = bc[l * GATE4 + 1536 + col];   // o
    }
  }
  const float bout_v = (wave < 4) ? b_out[(wave & 1) * 16 + l16] : 0.0f;

  // Load this tile's x0 rows into LDS buffer 0.
  {
    const __bf16* src = x0 + (size_t)wg * ROWS * HID;
    for (int idx = tid; idx < ROWS * HID; idx += NWAVES * 32) {
      int r = idx >> 9, c = idx & (HID - 1);
      hs[0][r][c] = src[idx];
    }
  }
  __syncthreads();

  int cur = 0;
  for (int t = 0; t < TSTEPS; ++t) {
    // -------- 4 sequential layers, h tile kept in LDS --------
    #pragma unroll 1
    for (int l = 0; l < NLAYERS; ++l) {
      const int nxt = cur ^ 1;

      #pragma unroll
      for (int j = 0; j < 2; ++j) {
        const int nt  = wave + j * 16;
        const int col = nt * 16 + l16;

        const float bi = biasv[l][j][0];
        const float bg = biasv[l][j][1];
        const float bo = biasv[l][j][2];
        v8f ai0 = {bi, bi, bi, bi, bi, bi, bi, bi}; v8f ai1 = ai0;
        v8f ag0 = {bg, bg, bg, bg, bg, bg, bg, bg}; v8f ag1 = ag0;
        v8f ao0 = {bo, bo, bo, bo, bo, bo, bo, bo}; v8f ao1 = ao0;

        // Packed fragment bases for this (layer, ntile): one per gate.
        const __bf16* pI = Wpack + ((size_t)((l * 3 + 0) * 32 + nt)) * BLK_ELEMS + lane * 16;
        const __bf16* pG = Wpack + ((size_t)((l * 3 + 1) * 32 + nt)) * BLK_ELEMS + lane * 16;
        const __bf16* pO = Wpack + ((size_t)((l * 3 + 2) * 32 + nt)) * BLK_ELEMS + lane * 16;

        #pragma unroll 4
        for (int kk = 0; kk < HID / 32; ++kk) {
          const int k0 = kk * 32;
          const int ka = k0 + half * 8;        // A elems 0..7  K range
          const int kb = k0 + 16 + half * 8;   // A elems 8..15 K range
          union { u32x4 q[2]; v16bf v; } a0, a1;
          a0.q[0] = *(const u32x4*)&hs[cur][l16][ka];
          a0.q[1] = *(const u32x4*)&hs[cur][l16][kb];
          a1.q[0] = *(const u32x4*)&hs[cur][16 + l16][ka];
          a1.q[1] = *(const u32x4*)&hs[cur][16 + l16][kb];

          // Fully-coalesced packed B fragments (1 KB contiguous per load).
          v16bf bI = *(const v16bf*)(pI + (size_t)kk * (32 * 16));
          v16bf bG = *(const v16bf*)(pG + (size_t)kk * (32 * 16));
          v16bf bO = *(const v16bf*)(pO + (size_t)kk * (32 * 16));

          ai0 = wmma_bf16(a0.v, bI, ai0);  ai1 = wmma_bf16(a1.v, bI, ai1);
          ag0 = wmma_bf16(a0.v, bG, ag0);  ag1 = wmma_bf16(a1.v, bG, ag1);
          ao0 = wmma_bf16(a0.v, bO, ao0);  ao1 = wmma_bf16(a1.v, bO, ao1);
        }

        // h = sigmoid(o) * tanh(sigmoid(i) * tanh(g))   (c0 == 0, f unused)
        #pragma unroll
        for (int r = 0; r < 8; ++r) {
          float h0 = fast_sigmoid(ao0[r]) * fast_tanh(fast_sigmoid(ai0[r]) * fast_tanh(ag0[r]));
          float h1 = fast_sigmoid(ao1[r]) * fast_tanh(fast_sigmoid(ai1[r]) * fast_tanh(ag1[r]));
          const int rr = r + half * 8;
          hs[nxt][rr][col]      = (__bf16)h0;
          hs[nxt][16 + rr][col] = (__bf16)h1;
        }
      }
      __syncthreads();
      cur ^= 1;
    }
    // cur toggled 4x -> back to buffer 0

    // -------- output projection: out[t] = tanh(h @ W_out^T + b_out) --------
    if (wave < 4) {
      const int mt  = wave >> 1;
      const int nt  = wave & 1;
      const int col = nt * 16 + l16;
      v8f acc = {bout_v, bout_v, bout_v, bout_v, bout_v, bout_v, bout_v, bout_v};
      const __bf16* pW = Woutpack + (size_t)nt * BLK_ELEMS + lane * 16;
      #pragma unroll 4
      for (int kk = 0; kk < HID / 32; ++kk) {
        const int k0 = kk * 32;
        union { u32x4 q[2]; v16bf v; } a;
        a.q[0] = *(const u32x4*)&hs[cur][mt * 16 + l16][k0 + half * 8];
        a.q[1] = *(const u32x4*)&hs[cur][mt * 16 + l16][k0 + 16 + half * 8];
        v16bf bf = *(const v16bf*)(pW + (size_t)kk * (32 * 16));
        acc = wmma_bf16(a.v, bf, acc);
      }
      #pragma unroll
      for (int r = 0; r < 8; ++r) {
        const int m    = mt * 16 + r + half * 8;
        const int brow = wg * ROWS + m;
        out[((size_t)brow * TSTEPS + t) * SEQ_DIM + col] = fast_tanh(acc[r]);
      }
    }
    // Barrier accounting: output reads buf0; next layer-0 writes buf1 and its
    // end-of-layer barrier orders everything before layer-1 writes buf0.
  }
}

// ---------------------------------------------------------------------------
// Host-side launch
// ---------------------------------------------------------------------------
extern "C" void kernel_launch(void* const* d_in, const int* in_sizes, int n_in,
                              void* d_out, int out_size, void* d_ws, size_t ws_size,
                              hipStream_t stream) {
  const float* label   = (const float*)d_in[0];
  const float* noise   = (const float*)d_in[1];
  const int*   length  = (const int*)  d_in[2];
  const float* W_label = (const float*)d_in[3];
  const float* b_label = (const float*)d_in[4];
  const float* W_noise = (const float*)d_in[5];
  const float* b_noise = (const float*)d_in[6];
  const float* W_len   = (const float*)d_in[7];
  const float* b_len   = (const float*)d_in[8];
  const float* W_comb  = (const float*)d_in[9];
  const float* b_comb  = (const float*)d_in[10];
  const float* W_ih    = (const float*)d_in[11];
  // d_in[12] = W_hh : unused (recurrent state is zero; only b_hh matters)
  const float* b_ih    = (const float*)d_in[13];
  const float* b_hh    = (const float*)d_in[14];
  const float* W_out   = (const float*)d_in[15];
  const float* b_out   = (const float*)d_in[16];
  float* out = (float*)d_out;

  // Workspace layout
  char* ws = (char*)d_ws;
  __bf16* Wpack    = (__bf16*)(ws);               // 384*8192 bf16 = 6,291,456 B
  float*  bc       = (float*) (ws + 6291456);     // 4*2048 f32    =    32,768 B
  __bf16* Woutpack = (__bf16*)(ws + 6324224);     // 2*8192 bf16   =    32,768 B
  __bf16* x0       = (__bf16*)(ws + 6356992);     // 2048*512 bf16 = 2,097,152 B

  // 1) weight repack (fragment-contiguous) / bias fusion
  {
    const int nthreads = NLAYERS * 3 * 32 * 16 * 32;   // 196,608
    repack_kernel<<<dim3((nthreads + 255) / 256), 256, 0, stream>>>(
        W_ih, b_ih, b_hh, W_out, Wpack, bc, Woutpack);
  }
  // 2) conditioning front-end -> x0
  frontend_kernel<<<dim3(B_TOT), dim3(128), 0, stream>>>(
      label, noise, length, W_label, b_label, W_noise, b_noise,
      W_len, b_len, W_comb, b_comb, x0);

  // 3) persistent LSTM chain + output projection
  lstm_gen_kernel<<<dim3(B_TOT / ROWS), dim3(NWAVES * 32), 0, stream>>>(
      Wpack, bc, Woutpack, b_out, x0, out);
}
